// HiddenCrfRnnLayer_84335977825015
// MI455X (gfx1250) — matrix operand order repaired and checked by reading, not compiled, not run on hardware
//
#include <hip/hip_runtime.h>
#include <hip/hip_bf16.h>

// ---------------------------------------------------------------------------
// Problem constants
// ---------------------------------------------------------------------------
#define HH 80
#define WW 80
#define NN 6400          // H*W pixels
#define CC 6             // channels / labels
#define NLAB 6
#define ROWS 36          // NLAB * CC batched rows
#define ROWS_PAD 48      // padded to 3 WMMA M-tiles of 16

static constexpr float INV2A = 1.0f / (2.0f * 8.0f  * 8.0f );   // bilateral spatial
static constexpr float INV2B = 1.0f / (2.0f * 0.15f * 0.15f);   // bilateral color
static constexpr float INV2G = 1.0f / (2.0f * 3.0f  * 3.0f );   // spatial-only

// ---------------------------------------------------------------------------
// WMMA types (gfx1250, wave32)
// ---------------------------------------------------------------------------
typedef __bf16 bf16x16 __attribute__((ext_vector_type(16)));
typedef float  f32x8   __attribute__((ext_vector_type(8)));

struct alignas(16) U4 { unsigned int x, y, z, w; };
union BFPack { U4 u[2]; bf16x16 v; };

// ---------------------------------------------------------------------------
// 1) Build row-normalized bilateral kernel Kb in bf16 (82 MB -> L2 resident).
//    Kb_norm[i,j] = exp(-d2p/2ta^2 - d2f/2tb^2) / norm_b[i]
//    One block per row i; row staged in LDS to normalize in a single pass.
// ---------------------------------------------------------------------------
__global__ void build_kb_kernel(const float* __restrict__ feat,
                                __hip_bfloat16* __restrict__ kbn) {
  __shared__ float rowbuf[NN];
  __shared__ float red[256];
  const int i  = blockIdx.x;
  const int yi = i / WW, xi = i % WW;
  const float f0 = feat[3 * i + 0];
  const float f1 = feat[3 * i + 1];
  const float f2 = feat[3 * i + 2];
  float s = 0.0f;
  for (int j = threadIdx.x; j < NN; j += 256) {
    const int yj = j / WW, xj = j % WW;
    const float dy = (float)(yi - yj), dx = (float)(xi - xj);
    const float g0 = feat[3 * j + 0] - f0;
    const float g1 = feat[3 * j + 1] - f1;
    const float g2 = feat[3 * j + 2] - f2;
    const float e = (dy * dy + dx * dx) * INV2A +
                    (g0 * g0 + g1 * g1 + g2 * g2) * INV2B;
    const float v = __expf(-e);
    rowbuf[j] = v;
    s += v;
  }
  red[threadIdx.x] = s;
  __syncthreads();
  for (int st = 128; st > 0; st >>= 1) {
    if (threadIdx.x < (unsigned)st) red[threadIdx.x] += red[threadIdx.x + st];
    __syncthreads();
  }
  const float inv = 1.0f / red[0];
  for (int j = threadIdx.x; j < NN; j += 256)
    kbn[(size_t)i * NN + j] = __float2bfloat16(rowbuf[j] * inv);
}

// ---------------------------------------------------------------------------
// 2) Spatial normalizer (separable): norm_s[i] = Sy[yi] * Sx[xi]
// ---------------------------------------------------------------------------
__global__ void norm_s_kernel(float* __restrict__ norms) {
  const int i = blockIdx.x * 256 + threadIdx.x;
  if (i >= NN) return;
  const int y = i / WW, x = i % WW;
  float sy = 0.0f, sx = 0.0f;
  for (int t = 0; t < HH; ++t) {
    const float d = (float)(y - t);
    sy += __expf(-d * d * INV2G);
  }
  for (int t = 0; t < WW; ++t) {
    const float d = (float)(x - t);
    sx += __expf(-d * d * INV2G);
  }
  norms[i] = sy * sx;
}

// ---------------------------------------------------------------------------
// 3) Pre-fold mixing matrices: M1 = comp @ Wsp, M2 = comp @ Wbl  (6x6 each)
// ---------------------------------------------------------------------------
__global__ void mats_kernel(const float* __restrict__ comp,
                            const float* __restrict__ wsp,
                            const float* __restrict__ wbl,
                            float* __restrict__ m1, float* __restrict__ m2) {
  const int t = threadIdx.x;
  if (t >= CC * CC) return;
  const int r = t / CC, c = t % CC;
  float a = 0.0f, b = 0.0f;
  for (int k = 0; k < CC; ++k) {
    a += comp[r * CC + k] * wsp[k * CC + c];
    b += comp[r * CC + k] * wbl[k * CC + c];
  }
  m1[t] = a;
  m2[t] = b;
}

// ---------------------------------------------------------------------------
// 4) q init: q[l][c][i] = unary[c][i] for every label l.
//    unary input layout: [1,H,W,C] -> flat (i*C + c)
// ---------------------------------------------------------------------------
__global__ void init_q_kernel(const float* __restrict__ unary,
                              float* __restrict__ q) {
  const int r = blockIdx.y;                      // 0..35
  const int i = blockIdx.x * 256 + threadIdx.x;  // 0..6399
  q[(size_t)r * NN + i] = unary[i * CC + (r % CC)];
}

// Zero the 12 padding rows (36..47) of the bf16 sm matrix once per call.
__global__ void zero_pad_kernel(__hip_bfloat16* __restrict__ smB) {
  const int idx = blockIdx.x * 256 + threadIdx.x;
  if (idx < (ROWS_PAD - ROWS) * NN)
    smB[(size_t)ROWS * NN + idx] = __float2bfloat16(0.0f);
}

// ---------------------------------------------------------------------------
// 5) Per-pixel, per-label channel softmax.  Writes f32 (for conv/ELBO) and
//    bf16 (for WMMA).
// ---------------------------------------------------------------------------
__global__ void softmax_kernel(const float* __restrict__ q,
                               float* __restrict__ smF,
                               __hip_bfloat16* __restrict__ smB) {
  const int l = blockIdx.y;
  const int i = blockIdx.x * 256 + threadIdx.x;
  float v[CC];
  float mx = -3.0e38f;
  for (int c = 0; c < CC; ++c) {
    v[c] = q[(size_t)(l * CC + c) * NN + i];
    mx = fmaxf(mx, v[c]);
  }
  float s = 0.0f;
  for (int c = 0; c < CC; ++c) { v[c] = __expf(v[c] - mx); s += v[c]; }
  const float inv = 1.0f / s;
  for (int c = 0; c < CC; ++c) {
    const float p = v[c] * inv;
    smF[(size_t)(l * CC + c) * NN + i] = p;
    smB[(size_t)(l * CC + c) * NN + i] = __float2bfloat16(p);
  }
}

// ---------------------------------------------------------------------------
// 6) Bilateral message via WMMA: bl[r, i] = sum_j smB[r, j] * Kb_norm[i, j]
//    (Kb symmetric + row-normalized => loading ROW i of kbn gives exactly the
//    normalized column the matmul needs, as contiguous 16-byte chunks.)
//    One wave per 16-column N-tile; 3 M-tiles (48 padded rows) per wave;
//    K swept in chunks of 32 with v_wmma_f32_16x16x32_bf16.
// ---------------------------------------------------------------------------
__global__ void bilateral_wmma_kernel(const __hip_bfloat16* __restrict__ smB,
                                      const __hip_bfloat16* __restrict__ kbn,
                                      float* __restrict__ bl) {
  const int lane  = threadIdx.x & 31;
  const int wave  = threadIdx.x >> 5;
  const int ntile = blockIdx.x * 4 + wave;   // 0..399
  const int nbase = ntile * 16;
  const int ln    = lane & 15;               // column within the N-tile
  const int hi    = (lane >> 4) & 1;         // half-wave selector

  f32x8 acc[3] = {};

  // B operand: lane holds K = kb + hi*8 + {0..7, 16..23} of (normalized) row nbase+ln
  const __hip_bfloat16* brow = kbn + (size_t)(nbase + ln) * NN + hi * 8;

  #pragma unroll 2
  for (int kb = 0; kb < NN; kb += 32) {
    BFPack b;
    b.u[0] = *(const U4*)(brow + kb);
    b.u[1] = *(const U4*)(brow + kb + 16);
    #pragma unroll
    for (int m = 0; m < 3; ++m) {
      // A operand: lane holds row m*16+ln, K = kb + hi*8 + {0..7, 16..23}
      const __hip_bfloat16* arow =
          smB + (size_t)(m * 16 + ln) * NN + kb + hi * 8;
      BFPack a;
      a.u[0] = *(const U4*)(arow);
      a.u[1] = *(const U4*)(arow + 16);
      acc[m] = __builtin_amdgcn_wmma_f32_16x16x32_bf16(
          false, a.v, false, b.v, (short)0, acc[m], false, false);
    }
  }

  // D layout: VGPR e holds rows e (lanes 0-15) / e+8 (lanes 16-31), col = ln
  #pragma unroll
  for (int m = 0; m < 3; ++m) {
    #pragma unroll
    for (int e = 0; e < 8; ++e) {
      const int r = m * 16 + e + hi * 8;
      if (r < ROWS) bl[(size_t)r * NN + nbase + ln] = acc[m][e];
    }
  }
}

// ---------------------------------------------------------------------------
// 7) Spatial message: exact separable full-width Gaussian (Ks is separable on
//    the grid).  conv_x then conv_y; conv_y also applies /norm_s.
// ---------------------------------------------------------------------------
__global__ void conv_x_kernel(const float* __restrict__ smF,
                              float* __restrict__ tmp) {
  __shared__ float row[WW];
  __shared__ float wt[WW];
  const int y = blockIdx.x, r = blockIdx.y, x = threadIdx.x;
  row[x] = smF[(size_t)r * NN + y * WW + x];
  wt[x]  = __expf(-(float)(x * x) * INV2G);
  __syncthreads();
  float s = 0.0f;
  for (int t = 0; t < WW; ++t) {
    const int d = x > t ? x - t : t - x;
    s += wt[d] * row[t];
  }
  tmp[(size_t)r * NN + y * WW + x] = s;
}

__global__ void conv_y_kernel(const float* __restrict__ tmp,
                              const float* __restrict__ norms,
                              float* __restrict__ sp) {
  __shared__ float col[HH];
  __shared__ float wt[HH];
  const int x = blockIdx.x, r = blockIdx.y, y = threadIdx.x;
  col[y] = tmp[(size_t)r * NN + y * WW + x];
  wt[y]  = __expf(-(float)(y * y) * INV2G);
  __syncthreads();
  float s = 0.0f;
  for (int t = 0; t < HH; ++t) {
    const int d = y > t ? y - t : t - y;
    s += wt[d] * col[t];
  }
  const int i = y * WW + x;
  sp[(size_t)r * NN + i] = s / norms[i];
}

// ---------------------------------------------------------------------------
// 8) q update: q[l][c][i] = unary[c][i] + LG[c][l]
//                         - sum_k (M1[c][k]*sp[l,k,i] + M2[c][k]*bl[l,k,i])
// ---------------------------------------------------------------------------
__global__ void combine_kernel(const float* __restrict__ unary,
                               const float* __restrict__ lg,
                               const float* __restrict__ m1,
                               const float* __restrict__ m2,
                               const float* __restrict__ sp,
                               const float* __restrict__ bl,
                               float* __restrict__ q) {
  const int l = blockIdx.y;
  const int i = blockIdx.x * 256 + threadIdx.x;
  float spv[CC], blv[CC];
  for (int k = 0; k < CC; ++k) {
    spv[k] = sp[(size_t)(l * CC + k) * NN + i];
    blv[k] = bl[(size_t)(l * CC + k) * NN + i];
  }
  for (int c = 0; c < CC; ++c) {
    float pw = 0.0f;
    for (int k = 0; k < CC; ++k)
      pw += m1[c * CC + k] * spv[k] + m2[c * CC + k] * blv[k];
    q[(size_t)(l * CC + c) * NN + i] =
        unary[i * CC + c] + lg[c * CC + l] - pw;
  }
}

// ---------------------------------------------------------------------------
// 9) ELBO: per-pixel contribution, deterministic two-stage tree reduction.
//    msg = Wsp@sp_raw + Wbl@bl_raw (no comp here, per reference).
// ---------------------------------------------------------------------------
__global__ void elbo_partial_kernel(const float* __restrict__ smF,
                                    const float* __restrict__ sp,
                                    const float* __restrict__ bl,
                                    const float* __restrict__ unary,
                                    const float* __restrict__ lg,
                                    const float* __restrict__ wsp,
                                    const float* __restrict__ wbl,
                                    float* __restrict__ partial) {
  __shared__ float red[128];
  const int l = blockIdx.y;
  const int i = blockIdx.x * 128 + threadIdx.x;   // 50*128 = 6400 exactly
  float sm[CC], spv[CC], blv[CC];
  for (int c = 0; c < CC; ++c) {
    sm[c]  = smF[(size_t)(l * CC + c) * NN + i];
    spv[c] = sp [(size_t)(l * CC + c) * NN + i];
    blv[c] = bl [(size_t)(l * CC + c) * NN + i];
  }
  float contrib = 0.0f;
  for (int c = 0; c < CC; ++c) {
    float msg = 0.0f;
    for (int k = 0; k < CC; ++k)
      msg += wsp[c * CC + k] * spv[k] + wbl[c * CC + k] * blv[k];
    const float s = sm[c];
    contrib += s * (unary[i * CC + c] + lg[c * CC + l] - msg)
               - s * __logf(s + 1e-10f);
  }
  red[threadIdx.x] = contrib;
  __syncthreads();
  for (int st = 64; st > 0; st >>= 1) {
    if (threadIdx.x < (unsigned)st) red[threadIdx.x] += red[threadIdx.x + st];
    __syncthreads();
  }
  if (threadIdx.x == 0) partial[l * 50 + blockIdx.x] = red[0];
}

__global__ void elbo_final_kernel(const float* __restrict__ partial,
                                  float* __restrict__ out) {
  __shared__ float red[64];
  const int l = blockIdx.x;
  const int t = threadIdx.x;
  red[t] = (t < 50) ? partial[l * 50 + t] : 0.0f;
  __syncthreads();
  for (int st = 32; st > 0; st >>= 1) {
    if (t < (unsigned)st) red[t] += red[t + st];
    __syncthreads();
  }
  if (t == 0) out[l] = red[0];
}

// ---------------------------------------------------------------------------
// Launch sequence
// ---------------------------------------------------------------------------
extern "C" void kernel_launch(void* const* d_in, const int* in_sizes, int n_in,
                              void* d_out, int out_size, void* d_ws,
                              size_t ws_size, hipStream_t stream) {
  (void)in_sizes; (void)n_in; (void)out_size; (void)ws_size;

  const float* unary = (const float*)d_in[0];   // [1,80,80,6] -> i*6+c
  const float* feat  = (const float*)d_in[1];   // [80,80,3]
  const float* comp  = (const float*)d_in[2];   // [6,6]
  const float* lg    = (const float*)d_in[3];   // [6,6]
  const float* wsp   = (const float*)d_in[4];   // [6,6]
  const float* wbl   = (const float*)d_in[5];   // [6,6]
  float* out = (float*)d_out;

  char* ws = (char*)d_ws;
  size_t off = 0;
  auto carve = [&](size_t bytes) -> void* {
    void* p = ws + off;
    off += (bytes + 255) & ~(size_t)255;
    return p;
  };
  __hip_bfloat16* kbn = (__hip_bfloat16*)carve((size_t)NN * NN * 2);      // 82 MB
  __hip_bfloat16* smB = (__hip_bfloat16*)carve((size_t)ROWS_PAD * NN * 2);
  float* smF     = (float*)carve((size_t)ROWS * NN * 4);
  float* q       = (float*)carve((size_t)ROWS * NN * 4);
  float* bl      = (float*)carve((size_t)ROWS * NN * 4);
  float* sp      = (float*)carve((size_t)ROWS * NN * 4);
  float* tmp     = (float*)carve((size_t)ROWS * NN * 4);
  float* norms   = (float*)carve((size_t)NN * 4);
  float* m1      = (float*)carve(CC * CC * 4);
  float* m2      = (float*)carve(CC * CC * 4);
  float* partial = (float*)carve(NLAB * 50 * 4);

  // One-time-per-call setup
  build_kb_kernel<<<NN, 256, 0, stream>>>(feat, kbn);
  norm_s_kernel<<<NN / 256, 256, 0, stream>>>(norms);
  mats_kernel<<<1, 64, 0, stream>>>(comp, wsp, wbl, m1, m2);
  init_q_kernel<<<dim3(NN / 256, ROWS), 256, 0, stream>>>(unary, q);
  zero_pad_kernel<<<((ROWS_PAD - ROWS) * NN + 255) / 256, 256, 0, stream>>>(smB);

  // Mean-field iterations (5 softmax+message passes; last q-update unused)
  for (int it = 0; it < 5; ++it) {
    softmax_kernel<<<dim3(NN / 256, NLAB), 256, 0, stream>>>(q, smF, smB);
    bilateral_wmma_kernel<<<100, 128, 0, stream>>>(smB, kbn, bl);
    conv_x_kernel<<<dim3(HH, ROWS), WW, 0, stream>>>(smF, tmp);
    conv_y_kernel<<<dim3(WW, ROWS), HH, 0, stream>>>(tmp, norms, sp);
    if (it < 4)
      combine_kernel<<<dim3(NN / 256, NLAB), 256, 0, stream>>>(
          unary, lg, m1, m2, sp, bl, q);
  }

  // ELBO from the last softmax + its message components
  elbo_partial_kernel<<<dim3(50, NLAB), 128, 0, stream>>>(
      smF, sp, bl, unary, lg, wsp, wbl, partial);
  elbo_final_kernel<<<NLAB, 64, 0, stream>>>(partial, out);
}
